// AtomEncoderLayer_4157528342954
// MI455X (gfx1250) — compile-verified, matrix operand order checked by reading, not run on hardware
//
#include <hip/hip_runtime.h>
#include <hip/hip_bf16.h>

#define B_   4
#define N_   32
#define D_   128
#define H_   8
#define NC_  27
#define DK_  16
#define DFF_ 512

typedef __attribute__((ext_vector_type(16))) _Float16 v16h;
typedef __attribute__((ext_vector_type(8)))  _Float16 v8h;
typedef __attribute__((ext_vector_type(4)))  _Float16 v4h;
typedef __attribute__((ext_vector_type(8)))  float    v8f;

__device__ __forceinline__ float wave_sum(float v) {
#pragma unroll
    for (int m = 16; m > 0; m >>= 1) v += __shfl_xor(v, m, 32);
    return v;
}

// WMMA f16 16x16x32 fragment load (A row-major; B from pre-transposed W^T).
// lane<16:  elems[0..7]=K kk..kk+7,   [8..15]=K kk+16..kk+23
// lane>=16: elems[0..7]=K kk+8..kk+15,[8..15]=K kk+24..kk+31
__device__ __forceinline__ v16h load_frag(const _Float16* base, int row, int pitch,
                                          int kk, int lane) {
    int half = lane >> 4;
    const _Float16* p = base + row * pitch + kk + 8 * half;
    v8h lo = *(const v8h*)p;
    v8h hi = *(const v8h*)(p + 16);
    v16h r;
#pragma unroll
    for (int j = 0; j < 8; ++j) { r[j] = lo[j]; r[j + 8] = hi[j]; }
    return r;
}

#define WMMA_F16(A, Bf, C) __builtin_amdgcn_wmma_f32_16x16x32_f16( \
    false, (A), false, (Bf), (short)0, (C), false, false)

// ---------------------------------------------------------------------------
// Prep: transpose + convert weights to f16 so WMMA B-fragments are contiguous.
// ---------------------------------------------------------------------------
__global__ __launch_bounds__(256) void prep_weights_kernel(
    const float* __restrict__ Wk, const float* __restrict__ Wv,
    const float* __restrict__ W1, const float* __restrict__ W2,
    _Float16* __restrict__ wkT, _Float16* __restrict__ wvT,
    _Float16* __restrict__ w1T, _Float16* __restrict__ w2T) {
    int tid = blockIdx.x * 256 + threadIdx.x;
    if (tid < 16384) {
        int n = tid >> 7, k = tid & 127;
        wkT[tid] = (_Float16)Wk[k * D_ + n];
        wvT[tid] = (_Float16)Wv[k * D_ + n];
    }
    if (tid < 65536) {
        { int n = tid >> 7, k = tid & 127; w1T[tid] = (_Float16)W1[k * DFF_ + n]; }
        { int n = tid >> 9, k = tid & 511; w2T[tid] = (_Float16)W2[k * D_ + n]; }
    }
}

// ---------------------------------------------------------------------------
// One workgroup per (b, i): streaming K/V projection (WMMA) + single-query
// online-softmax attention over L = 27*32 keys. Writes y[b,i,:] (pre-FFN).
// ---------------------------------------------------------------------------
__global__ __launch_bounds__(256) void atom_attn_kernel(
    const float* __restrict__ x, const int* __restrict__ mask,
    const float* __restrict__ coords, const float* __restrict__ Wp,
    const float* __restrict__ bp, const float* __restrict__ Wq,
    const float* __restrict__ bq, const float* __restrict__ bk,
    const float* __restrict__ bv, const float* __restrict__ Wo,
    const float* __restrict__ bo, const float* __restrict__ g1,
    const float* __restrict__ b1, const _Float16* __restrict__ wkT,
    const _Float16* __restrict__ wvT, float* __restrict__ yws) {
    __shared__ _Float16 sHn[N_ * D_];     // LN'd a_i tile (f16)   8 KB
    __shared__ float    sK[N_ * D_];      // K tile (+bk)         16 KB
    __shared__ float    sV[N_ * D_];      // V tile (+bv)         16 KB
    __shared__ float    sS[N_ * H_];      // scores
    __shared__ float    sE[N_ * H_];      // exp weights
    __shared__ float    sRaw[D_], sHq[D_], sQ[D_], sAtt[D_];
    __shared__ float    sM[H_], sScale[H_], sMnew[H_];
    __shared__ float    sMu2[2];
    __shared__ int      sMask[N_];

    const int tid = threadIdx.x;
    const int wv = tid >> 5, lane = tid & 31;
    const int b = blockIdx.x >> 5, qi = blockIdx.x & 31;
    const int d0 = lane * 4;              // 4 contiguous dims per lane

    // ---- query row: a_i[flat=i] = x[b,qi]+bp (coord term cancels) ----
    if (tid < D_) sRaw[tid] = x[(b * N_ + qi) * D_ + tid] + bp[tid];
    if (tid < N_) sMask[tid] = mask[b * N_ + tid];
    if (tid < H_) sM[tid] = -3e38f;
    __syncthreads();
    if (wv == 0) {
        float s = 0.f, s2 = 0.f;
#pragma unroll
        for (int j = 0; j < 4; ++j) { float v = sRaw[d0 + j]; s += v; s2 += v * v; }
        s = wave_sum(s); s2 = wave_sum(s2);
        if (lane == 0) {
            float mu = s * (1.f / D_);
            sMu2[0] = mu;
            sMu2[1] = rsqrtf(s2 * (1.f / D_) - mu * mu + 1e-5f);
        }
    }
    __syncthreads();
    if (tid < D_) sHq[tid] = (sRaw[tid] - sMu2[0]) * sMu2[1] * g1[tid] + b1[tid];
    __syncthreads();
    if (tid < D_) {
        float acc = bq[tid];
#pragma unroll 4
        for (int k = 0; k < D_; ++k) acc += sHq[k] * Wq[k * D_ + tid];
        sQ[tid] = acc;
    }

    // ---- loop-invariant hoists ----
    // phase-1 row data: x rows + bp, Wp columns, LN affine (per-lane float4)
    float bpv[4], wpx[4], wpy[4], wpz[4], g1v[4], b1v[4];
    {
        const float4 t0 = *(const float4*)&bp[d0];
        const float4 t1 = *(const float4*)&Wp[d0];
        const float4 t2 = *(const float4*)&Wp[D_ + d0];
        const float4 t3 = *(const float4*)&Wp[2 * D_ + d0];
        const float4 t4 = *(const float4*)&g1[d0];
        const float4 t5 = *(const float4*)&b1[d0];
        bpv[0]=t0.x; bpv[1]=t0.y; bpv[2]=t0.z; bpv[3]=t0.w;
        wpx[0]=t1.x; wpx[1]=t1.y; wpx[2]=t1.z; wpx[3]=t1.w;
        wpy[0]=t2.x; wpy[1]=t2.y; wpy[2]=t2.z; wpy[3]=t2.w;
        wpz[0]=t3.x; wpz[1]=t3.y; wpz[2]=t3.z; wpz[3]=t3.w;
        g1v[0]=t4.x; g1v[1]=t4.y; g1v[2]=t4.z; g1v[3]=t4.w;
        b1v[0]=t5.x; b1v[1]=t5.y; b1v[2]=t5.z; b1v[3]=t5.w;
    }
    float xr[4][4];                       // (x[b, wv*4+rr, d0..d0+3] + bp)
#pragma unroll
    for (int rr = 0; rr < 4; ++rr) {
        int n = wv * 4 + rr;
        const float4 xv = *(const float4*)&x[(b * N_ + n) * D_ + d0];
        xr[rr][0] = xv.x + bpv[0]; xr[rr][1] = xv.y + bpv[1];
        xr[rr][2] = xv.z + bpv[2]; xr[rr][3] = xv.w + bpv[3];
    }
    // phase-2 weight fragments (loop invariant; nt = wv)
    v16h bKf[4], bVf[4];
    {
        int brow = wv * 16 + (lane & 15);
#pragma unroll
        for (int ks = 0; ks < 4; ++ks) {
            bKf[ks] = load_frag(wkT, brow, D_, ks * 32, lane);
            bVf[ks] = load_frag(wvT, brow, D_, ks * 32, lane);
        }
    }
    const int colw = wv * 16 + (lane & 15);
    const float bbk = bk[colw], bbv = bv[colw];

    float accR = 0.f, SR = 0.f;           // online-softmax state (tid<128)

    for (int c = 0; c < NC_; ++c) {
        __syncthreads();
        // ---- phase 1: a_i tile rows + LayerNorm -> sHn (f16) ----
        {
            const float* cq = &coords[((b * NC_ + c) * N_ + qi) * 3];
            float qx = cq[0], qy = cq[1], qz = cq[2];
#pragma unroll
            for (int rr = 0; rr < 4; ++rr) {
                int n = wv * 4 + rr;
                const float* cn = &coords[((b * NC_ + c) * N_ + n) * 3];
                float dx = cn[0] - qx, dy = cn[1] - qy, dz = cn[2] - qz;
                float vals[4], s = 0.f, s2 = 0.f;
#pragma unroll
                for (int j = 0; j < 4; ++j) {
                    float v = xr[rr][j] + dx * wpx[j] + dy * wpy[j] + dz * wpz[j];
                    vals[j] = v; s += v; s2 += v * v;
                }
                s = wave_sum(s); s2 = wave_sum(s2);
                float mu = s * (1.f / D_);
                float rstd = rsqrtf(s2 * (1.f / D_) - mu * mu + 1e-5f);
                v4h hv;
#pragma unroll
                for (int j = 0; j < 4; ++j)
                    hv[j] = (_Float16)((vals[j] - mu) * rstd * g1v[j] + b1v[j]);
                *(v4h*)&sHn[n * D_ + d0] = hv;
            }
        }
        __syncthreads();
        // ---- phase 2: K,V = hn @ Wk/Wv (batched fragment loads + 16 WMMA) ----
        {
            v16h a0[4], a1[4];
#pragma unroll
            for (int ks = 0; ks < 4; ++ks)
                a0[ks] = load_frag(sHn, (lane & 15), D_, ks * 32, lane);
#pragma unroll
            for (int ks = 0; ks < 4; ++ks)
                a1[ks] = load_frag(sHn, 16 + (lane & 15), D_, ks * 32, lane);
            v8f k0 = {}, v0 = {}, k1 = {}, v1 = {};
#pragma unroll
            for (int ks = 0; ks < 4; ++ks) {
                k0 = WMMA_F16(a0[ks], bKf[ks], k0);
                v0 = WMMA_F16(a0[ks], bVf[ks], v0);
                k1 = WMMA_F16(a1[ks], bKf[ks], k1);
                v1 = WMMA_F16(a1[ks], bVf[ks], v1);
            }
            int rb = (lane >> 4) * 8;
#pragma unroll
            for (int r = 0; r < 8; ++r) {
                sK[(rb + r) * D_ + colw]      = k0[r] + bbk;
                sK[(16 + rb + r) * D_ + colw] = k1[r] + bbk;
                sV[(rb + r) * D_ + colw]      = v0[r] + bbv;
                sV[(16 + rb + r) * D_ + colw] = v1[r] + bbv;
            }
        }
        __syncthreads();
        // ---- phase 3: scores s[row,h] = 0.25 * q_h . K[row,h] (masked) ----
        {
            int row = tid >> 3, h = tid & 7;
            float sc;
            if (sMask[row] == 0) sc = -3e38f;
            else {
                float dotv = 0.f;
#pragma unroll
                for (int d = 0; d < DK_; ++d)
                    dotv += sQ[h * DK_ + d] * sK[row * D_ + h * DK_ + d];
                sc = 0.25f * dotv;
            }
            sS[row * H_ + h] = sc;
        }
        __syncthreads();
        // ---- phase 4: per-head running max / rescale factor ----
        if (tid < H_) {
            float tm = -3e38f;
            for (int r2 = 0; r2 < N_; ++r2) tm = fmaxf(tm, sS[r2 * H_ + tid]);
            float mo = sM[tid];
            float mn = fmaxf(mo, tm);
            sMnew[tid] = mn;
            sScale[tid] = __expf(mo - mn);   // exp(-huge)=0 on first tile
            sM[tid] = mn;
        }
        __syncthreads();
        // ---- phase 5: exp weights ----
        {
            int row = tid >> 3, h = tid & 7;
            float s = sS[row * H_ + h];
            sE[row * H_ + h] = (s <= -1e38f) ? 0.f : __expf(s - sMnew[h]);
        }
        __syncthreads();
        // ---- phase 6: accumulate numerator / denominator ----
        if (tid < D_) {
            int h = tid >> 4, d = tid & 15;
            float sc = sScale[h];
            float es = 0.f, av = 0.f;
#pragma unroll 4
            for (int r2 = 0; r2 < N_; ++r2) {
                float e = sE[r2 * H_ + h];
                es += e;
                av += e * sV[r2 * D_ + h * DK_ + d];
            }
            accR = accR * sc + av;
            SR = SR * sc + es;
        }
        __syncthreads();
    }

    if (tid < D_) sAtt[tid] = (SR > 0.f) ? accR / SR : 0.f;
    __syncthreads();
    // ---- output row: y[b,i] = (x[b,i]+bp) + att @ Wo + bo ----
    if (tid < D_) {
        float o = x[(b * N_ + qi) * D_ + tid] + bp[tid] + bo[tid];
#pragma unroll 4
        for (int k = 0; k < D_; ++k) o += sAtt[k] * Wo[k * D_ + tid];
        yws[(b * N_ + qi) * D_ + tid] = o;
    }
}

// ---------------------------------------------------------------------------
// FFN: out = y + relu(LN(y) @ W1 + bf1) @ W2 + bf2 ; one WG per 16 rows.
// ---------------------------------------------------------------------------
__global__ __launch_bounds__(256) void atom_ffn_kernel(
    const float* __restrict__ yws, const float* __restrict__ g2,
    const float* __restrict__ b2, const float* __restrict__ bf1,
    const float* __restrict__ bf2, const _Float16* __restrict__ w1T,
    const _Float16* __restrict__ w2T, float* __restrict__ out) {
    __shared__ _Float16 sA[16 * D_];      // LN(y) f16
    __shared__ _Float16 sH[16 * DFF_];    // hidden f16
    const int tid = threadIdx.x, wv = tid >> 5, lane = tid & 31;
    const int r0 = blockIdx.x * 16;
    const int d0 = lane * 4;

    // LN (2 rows per wave, float4 loads, b64 f16 stores)
    {
        const float4 t4 = *(const float4*)&g2[d0];
        const float4 t5 = *(const float4*)&b2[d0];
        float g2v[4] = {t4.x, t4.y, t4.z, t4.w};
        float b2v[4] = {t5.x, t5.y, t5.z, t5.w};
#pragma unroll
        for (int rr = 0; rr < 2; ++rr) {
            int row = wv * 2 + rr;
            const float4 yv = *(const float4*)&yws[(r0 + row) * D_ + d0];
            float vals[4] = {yv.x, yv.y, yv.z, yv.w};
            float s = 0.f, s2 = 0.f;
#pragma unroll
            for (int j = 0; j < 4; ++j) { s += vals[j]; s2 += vals[j] * vals[j]; }
            s = wave_sum(s); s2 = wave_sum(s2);
            float mu = s * (1.f / D_);
            float rstd = rsqrtf(s2 * (1.f / D_) - mu * mu + 1e-5f);
            v4h hv;
#pragma unroll
            for (int j = 0; j < 4; ++j)
                hv[j] = (_Float16)((vals[j] - mu) * rstd * g2v[j] + b2v[j]);
            *(v4h*)&sA[row * D_ + d0] = hv;
        }
    }
    __syncthreads();
    // GEMM1 [16x128]@[128x512] + relu (A fragments shared across 4 N-tiles)
    {
        v16h aF[4];
#pragma unroll
        for (int ks = 0; ks < 4; ++ks)
            aF[ks] = load_frag(sA, lane & 15, D_, ks * 32, lane);
#pragma unroll
        for (int t = 0; t < 4; ++t) {
            int nt = wv * 4 + t;          // 0..31
            int brow = nt * 16 + (lane & 15);
            v16h bF[4];
#pragma unroll
            for (int ks = 0; ks < 4; ++ks)
                bF[ks] = load_frag(w1T, brow, D_, ks * 32, lane);
            v8f acc = {};
#pragma unroll
            for (int ks = 0; ks < 4; ++ks) acc = WMMA_F16(aF[ks], bF[ks], acc);
            int col = nt * 16 + (lane & 15);
            float bb = bf1[col];
            int rbase = (lane >> 4) * 8;
#pragma unroll
            for (int r = 0; r < 8; ++r) {
                float v = acc[r] + bb;
                sH[(rbase + r) * DFF_ + col] = (_Float16)(v > 0.f ? v : 0.f);
            }
        }
    }
    __syncthreads();
    // GEMM2 [16x512]@[512x128] + residual (k-steps chunked 4-deep for ILP)
    {
        int brow = wv * 16 + (lane & 15);
        v8f acc = {};
#pragma unroll
        for (int kb = 0; kb < 4; ++kb) {
            v16h aF2[4], bF2[4];
#pragma unroll
            for (int j = 0; j < 4; ++j)
                aF2[j] = load_frag(sH, lane & 15, DFF_, (kb * 4 + j) * 32, lane);
#pragma unroll
            for (int j = 0; j < 4; ++j)
                bF2[j] = load_frag(w2T, brow, DFF_, (kb * 4 + j) * 32, lane);
#pragma unroll
            for (int j = 0; j < 4; ++j) acc = WMMA_F16(aF2[j], bF2[j], acc);
        }
        int col = wv * 16 + (lane & 15);
        float bb = bf2[col];
        int rbase = (lane >> 4) * 8;
#pragma unroll
        for (int r = 0; r < 8; ++r) {
            int row = rbase + r;
            out[(r0 + row) * D_ + col] = yws[(r0 + row) * D_ + col] + acc[r] + bb;
        }
    }
}

extern "C" void kernel_launch(void* const* d_in, const int* in_sizes, int n_in,
                              void* d_out, int out_size, void* d_ws, size_t ws_size,
                              hipStream_t stream) {
    const float* x      = (const float*)d_in[0];
    const int*   mask   = (const int*)d_in[1];
    const float* coords = (const float*)d_in[2];
    const float* Wp     = (const float*)d_in[3];
    const float* bp     = (const float*)d_in[4];
    const float* Wq     = (const float*)d_in[5];
    const float* bq     = (const float*)d_in[6];
    const float* Wk     = (const float*)d_in[7];
    const float* bk     = (const float*)d_in[8];
    const float* Wv     = (const float*)d_in[9];
    const float* bv     = (const float*)d_in[10];
    const float* Wo     = (const float*)d_in[11];
    const float* bo     = (const float*)d_in[12];
    const float* g1     = (const float*)d_in[13];
    const float* b1     = (const float*)d_in[14];
    const float* g2     = (const float*)d_in[15];
    const float* b2     = (const float*)d_in[16];
    const float* W1     = (const float*)d_in[17];
    const float* bf1    = (const float*)d_in[18];
    const float* W2     = (const float*)d_in[19];
    const float* bf2    = (const float*)d_in[20];
    float* out = (float*)d_out;

    _Float16* wkT = (_Float16*)d_ws;          // 128*128
    _Float16* wvT = wkT + 16384;              // 128*128
    _Float16* w1T = wvT + 16384;              // 512*128
    _Float16* w2T = w1T + 65536;              // 128*512
    float*    yws = (float*)(w2T + 65536);    // B*N*D

    prep_weights_kernel<<<dim3(256), dim3(256), 0, stream>>>(
        Wk, Wv, W1, W2, wkT, wvT, w1T, w2T);
    atom_attn_kernel<<<dim3(B_ * N_), dim3(256), 0, stream>>>(
        x, mask, coords, Wp, bp, Wq, bq, bk, bv, Wo, bo, g1, b1, wkT, wvT, yws);
    atom_ffn_kernel<<<dim3((B_ * N_) / 16), dim3(256), 0, stream>>>(
        yws, g2, b2, bf1, bf2, w1T, w2T, out);
}
